// MultiHeadSelfAttention_21010980012533
// MI455X (gfx1250) — matrix operand order repaired and checked by reading, not compiled, not run on hardware
//
#include <hip/hip_runtime.h>
#include <hip/hip_bf16.h>

// MI455X / gfx1250, wave32. Fused MHA: bf16 WMMA GEMMs + flash attention
// with TDM (tensor_load_to_lds) double-buffered K tiles.

typedef __attribute__((ext_vector_type(16))) __bf16 v16bf;
typedef __attribute__((ext_vector_type(8)))  float  v8f;
typedef __attribute__((ext_vector_type(4)))  unsigned int v4u;
typedef __attribute__((ext_vector_type(8)))  int    v8i_t;
typedef __attribute__((ext_vector_type(4)))  int    v4i_t;

#define BB 4
#define LL 2048
#define EE 768
#define HH 12
#define DD 64

// float -> bf16 with round-to-nearest-even
__device__ __forceinline__ __bf16 f2bf(float f) {
    union { float f; unsigned u; } a; a.f = f;
    unsigned r = (a.u + 0x7FFFu + ((a.u >> 16) & 1u)) >> 16;
    union { unsigned short s; __bf16 b; } o; o.s = (unsigned short)r;
    return o.b;
}

// A-matrix (16-bit, 16x32) k-offset for VGPR slot v, lane-half hi (contiguous: k = 16*hi + 2*v)
__device__ __forceinline__ int ak_off(int v, int hi) {
    return 2 * v + 16 * hi;
}

__device__ __forceinline__ v8f wmma_bf16(v16bf a, v16bf b, v8f c) {
    return __builtin_amdgcn_wmma_f32_16x16x32_bf16(false, a, false, b, (short)0, c, false, false);
}

// ---------------------------------------------------------------------------
// TDM: load a 64x64 bf16 tile (row-major, row stride 64 elems) from global
// into LDS with 16-half padding per row (LDS row pitch = 80 halfs = 160B).
// D# built per CDNA5 ISA 8.3/8.4:
//   group0: count=1 | lds_addr | global_addr | type=2
//   group1: data_size=1(2B) | pad_enable, pad_interval=4(32 dw), pad_amount=7(8 dw)
//           tensor_dim0=64, tensor_dim1=64, tile_dim0=64, tile_dim1=64,
//           tensor_dim0_stride=64
// ---------------------------------------------------------------------------
__device__ __forceinline__ void tdm_load_k_tile_1250(const __bf16* gptr, unsigned lds_off) {
    unsigned long long ga = (unsigned long long)(uintptr_t)gptr;
    v4u g0;
    g0[0] = 1u;                                          // count=1 (valid user D#)
    g0[1] = lds_off;                                     // lds_addr (bytes)
    g0[2] = (unsigned)(ga & 0xFFFFFFFFu);                // global_addr[31:0]
    g0[3] = (unsigned)((ga >> 32) & 0x01FFFFFFu)         // global_addr[56:32]
          | (2u << 30);                                  // type=2 ("image")
    v8i_t g1;
    g1[0] = (int)((1u << 16)                             // data_size=1 -> 2 bytes
                | (1u << 20)                             // pad_enable
                | (4u << 22)                             // pad_interval: 32 DWORDs
                | (7u << 25));                           // pad_amount:   8 DWORDs
    g1[1] = (int)(64u << 16);                            // tensor_dim0[15:0]=64
    g1[2] = (int)(64u << 16);                            // tensor_dim1[15:0]=64
    g1[3] = (int)(64u << 16);                            // tile_dim0=64
    g1[4] = (int)(64u);                                  // tile_dim1=64, tile_dim2=0
    g1[5] = (int)(64u);                                  // tensor_dim0_stride=64
    g1[6] = 0;
    g1[7] = 0;
    v4i_t z4 = {0, 0, 0, 0};
#if defined(__clang_major__) && (__clang_major__ >= 23)
    v8i_t z8 = {0, 0, 0, 0, 0, 0, 0, 0};
    __builtin_amdgcn_tensor_load_to_lds(g0, g1, z4, z4, z8, 0);
#else
    __builtin_amdgcn_tensor_load_to_lds(g0, g1, z4, z4, 0);
#endif
}

// ---------------------------------------------------------------------------
// Kernel 1: QKV projection GEMM.  M=B*L=8192, N=E=768, K=E=768.
// grid = (M/64, N/64, 3) ; block = 128 (4 waves). z selects Wq/Wk/Wv.
// Output written as bf16 in [B*H][L][D] layout for the attention kernel.
// ---------------------------------------------------------------------------
__global__ __launch_bounds__(128) void qkv_gemm_1250(
    const float* __restrict__ x,
    const float* __restrict__ Wq, const float* __restrict__ Wk, const float* __restrict__ Wv,
    const float* __restrict__ bq, const float* __restrict__ bk, const float* __restrict__ bv,
    __bf16* __restrict__ Qb, __bf16* __restrict__ Kb, __bf16* __restrict__ Vb)
{
    const int z = blockIdx.z;
    const float* __restrict__ W    = (z == 0) ? Wq : ((z == 1) ? Wk : Wv);
    const float* __restrict__ bias = (z == 0) ? bq : ((z == 1) ? bk : bv);
    __bf16* __restrict__ Out       = (z == 0) ? Qb : ((z == 1) ? Kb : Vb);

    __shared__ __align__(16) __bf16 Xs[64][72];   // [m][k]  (x tile, bf16)
    __shared__ __align__(16) __bf16 Wt[64][72];   // [n][k]  (W tile transposed)

    const int tid  = threadIdx.x;
    const int wave = tid >> 5;
    const int lane = tid & 31;
    const int m16  = lane & 15;
    const int hi   = (lane >> 4) & 1;
    const int r0   = blockIdx.x * 64;   // row block in M
    const int n0   = blockIdx.y * 64;   // col block in N

    v8f acc[4] = {};

    for (int k0 = 0; k0 < EE; k0 += 64) {
        // stage x tile (f32 -> bf16)
        for (int i = tid; i < 64 * 64; i += 128) {
            int r = i >> 6, c = i & 63;
            Xs[r][c] = f2bf(x[(size_t)(r0 + r) * EE + (k0 + c)]);
        }
        // stage W tile transposed: Wt[n][k] = W[k0+k][n0+n]
        for (int i = tid; i < 64 * 64; i += 128) {
            int kk = i >> 6, nn = i & 63;
            Wt[nn][kk] = f2bf(W[(size_t)(k0 + kk) * EE + (n0 + nn)]);
        }
        __syncthreads();

        #pragma unroll
        for (int ks = 0; ks < 2; ++ks) {           // two K=32 WMMA steps
            v16bf a;
            #pragma unroll
            for (int v = 0; v < 8; ++v) {
                int k = ks * 32 + ak_off(v, hi);
                a[2 * v]     = Xs[wave * 16 + m16][k];
                a[2 * v + 1] = Xs[wave * 16 + m16][k + 1];
            }
            #pragma unroll
            for (int t = 0; t < 4; ++t) {
                v16bf bfr;
                #pragma unroll
                for (int v = 0; v < 8; ++v) {
                    int k = ks * 32 + 2 * v + 16 * hi;
                    bfr[2 * v]     = Wt[t * 16 + m16][k];
                    bfr[2 * v + 1] = Wt[t * 16 + m16][k + 1];
                }
                acc[t] = wmma_bf16(a, bfr, acc[t]);
            }
        }
        __syncthreads();
    }

    // bias add + scatter store into [B*H][L][D] bf16
    #pragma unroll
    for (int t = 0; t < 4; ++t) {
        int e = n0 + t * 16 + m16;
        float be = bias[e];
        int h = e >> 6, d = e & 63;
        #pragma unroll
        for (int v = 0; v < 8; ++v) {
            int r = r0 + wave * 16 + v + 8 * hi;   // row in [0, B*L)
            int b = r >> 11, l = r & (LL - 1);
            size_t off = ((size_t)(b * HH + h) * LL + l) * DD + d;
            Out[off] = f2bf(acc[t][v] + be);
        }
    }
}

// ---------------------------------------------------------------------------
// Kernel 2: flash attention, causal.  grid = (L/64, B*H); block = 128.
// Each wave owns 16 query rows; workgroup owns a 64-query block.
// K tiles arrive via TDM (double buffered, TENSORcnt-tracked); V is staged
// transposed by the waves; P reshaped C->A through per-wave LDS.
// ---------------------------------------------------------------------------
__global__ __launch_bounds__(128) void attn_flash_1250(
    const __bf16* __restrict__ Qb, const __bf16* __restrict__ Kb,
    const __bf16* __restrict__ Vb, float* __restrict__ y)
{
    __shared__ __align__(16) __bf16 Ks[2][64][80];   // [buf][key][d] (TDM-padded rows)
    __shared__ __align__(16) __bf16 Vt[64][80];      // [d][key]   (transposed)
    __shared__ __align__(16) __bf16 Ps[4][16][72];   // per-wave P staging [m][key]

    const int tid  = threadIdx.x;
    const int wave = tid >> 5;
    const int lane = tid & 31;
    const int m16  = lane & 15;
    const int hi   = (lane >> 4) & 1;
    const int qb   = blockIdx.x;       // query block index
    const int bh   = blockIdx.y;       // fused batch*head
    const int q0   = qb * 64;

    const size_t hbase = (size_t)bh * LL * DD;
    const __bf16* __restrict__ Qh = Qb + hbase;
    const __bf16* __restrict__ Kh = Kb + hbase;
    const __bf16* __restrict__ Vh = Vb + hbase;

    // Q fragments for this wave's 16 rows (K-dim = D = 64 -> 2 fragments)
    v16bf qf[2];
    #pragma unroll
    for (int f = 0; f < 2; ++f) {
        int l = q0 + wave * 16 + m16;
        #pragma unroll
        for (int v = 0; v < 8; ++v) {
            int d = f * 32 + ak_off(v, hi);
            qf[f][2 * v]     = Qh[(size_t)l * DD + d];
            qf[f][2 * v + 1] = Qh[(size_t)l * DD + d + 1];
        }
    }

    float mrun[8], lrun[8];
    v8f acc[4] = {};
    #pragma unroll
    for (int v = 0; v < 8; ++v) { mrun[v] = -1e30f; lrun[v] = 0.0f; }

    // kick off TDM for K tile 0 into buffer 0 (wave 0 only; EXEC all-ones here)
    if (wave == 0) {
        tdm_load_k_tile_1250(Kh, (unsigned)(uintptr_t)(void*)&Ks[0][0][0]);
    }

    for (int kb = 0; kb <= qb; ++kb) {
        const int buf = kb & 1;

        // ---- TDM: prefetch K tile kb+1 into the other buffer, then wait ----
        if (wave == 0) {
            if (kb < qb) {
                tdm_load_k_tile_1250(Kh + (size_t)(kb + 1) * 64 * DD,
                                     (unsigned)(uintptr_t)(void*)&Ks[buf ^ 1][0][0]);
                __builtin_amdgcn_s_wait_tensorcnt(1);   // in-order: tile kb done
            } else {
                __builtin_amdgcn_s_wait_tensorcnt(0);
            }
        }

        // ---- stage V tile transposed [d][key] (all waves) ----
        {
            const uint4* Vg = (const uint4*)(Vh + (size_t)kb * 64 * DD);
            for (int i = tid; i < 64 * 8; i += 128) {      // 8 halfs per uint4
                int row = i >> 3, c8 = (i & 7) * 8;
                union { uint4 q; __bf16 h[8]; } u; u.q = Vg[i];
                #pragma unroll
                for (int j = 0; j < 8; ++j) Vt[c8 + j][row] = u.h[j];
            }
            if (kb < qb)  // prefetch next V tile (global_prefetch_b8)
                __builtin_prefetch((const void*)(Vh + (size_t)(kb + 1) * 64 * DD), 0, 0);
        }
        __syncthreads();

        // ---- S = Q * K^T   (16x64 per wave, f32 accum) ----
        v8f s[4] = {};
        #pragma unroll
        for (int f = 0; f < 2; ++f) {
            #pragma unroll
            for (int t = 0; t < 4; ++t) {
                v16bf bfr;
                #pragma unroll
                for (int v = 0; v < 8; ++v) {
                    int k = f * 32 + 2 * v + 16 * hi;          // d index
                    bfr[2 * v]     = Ks[buf][t * 16 + m16][k]; // = K[key=n][d=k]
                    bfr[2 * v + 1] = Ks[buf][t * 16 + m16][k + 1];
                }
                s[t] = wmma_bf16(qf[f], bfr, s[t]);
            }
        }

        // ---- scale, causal mask (diag block only), row max ----
        const bool diag = (kb == qb);
        float rmax[8];
        #pragma unroll
        for (int v = 0; v < 8; ++v) rmax[v] = -1e30f;
        #pragma unroll
        for (int t = 0; t < 4; ++t) {
            int key = kb * 64 + t * 16 + m16;
            #pragma unroll
            for (int v = 0; v < 8; ++v) {
                float sv = s[t][v] * 0.125f;                 // 1/sqrt(64)
                if (diag) {
                    int qrow = q0 + wave * 16 + v + 8 * hi;
                    if (key > qrow) sv = -1e30f;
                }
                s[t][v] = sv;
                rmax[v] = fmaxf(rmax[v], sv);
            }
        }
        #pragma unroll
        for (int v = 0; v < 8; ++v) {                        // reduce over 16 lanes
            float r = rmax[v];
            r = fmaxf(r, __shfl_xor(r, 1, 32));
            r = fmaxf(r, __shfl_xor(r, 2, 32));
            r = fmaxf(r, __shfl_xor(r, 4, 32));
            r = fmaxf(r, __shfl_xor(r, 8, 32));
            rmax[v] = r;
        }

        // ---- online softmax update ----
        float alpha[8], rsum[8];
        #pragma unroll
        for (int v = 0; v < 8; ++v) {
            float mnew = fmaxf(mrun[v], rmax[v]);
            alpha[v] = __expf(mrun[v] - mnew);
            mrun[v]  = mnew;
            rsum[v]  = 0.0f;
        }
        #pragma unroll
        for (int t = 0; t < 4; ++t) {
            #pragma unroll
            for (int v = 0; v < 8; ++v) {
                float p = __expf(s[t][v] - mrun[v]);
                rsum[v] += p;
                Ps[wave][v + 8 * hi][t * 16 + m16] = f2bf(p);
            }
        }
        #pragma unroll
        for (int v = 0; v < 8; ++v) {
            float r = rsum[v];
            r += __shfl_xor(r, 1, 32);
            r += __shfl_xor(r, 2, 32);
            r += __shfl_xor(r, 4, 32);
            r += __shfl_xor(r, 8, 32);
            lrun[v] = lrun[v] * alpha[v] + r;
        }
        #pragma unroll
        for (int t = 0; t < 4; ++t)
            #pragma unroll
            for (int v = 0; v < 8; ++v)
                acc[t][v] *= alpha[v];

        // ---- O += P * V  (per-wave LDS round-trip to reshape C -> A frag) ----
        __builtin_amdgcn_wave_barrier();
        #pragma unroll
        for (int f = 0; f < 2; ++f) {
            v16bf pa;
            #pragma unroll
            for (int v = 0; v < 8; ++v) {
                int k = f * 32 + ak_off(v, hi);              // key index in block
                pa[2 * v]     = Ps[wave][m16][k];
                pa[2 * v + 1] = Ps[wave][m16][k + 1];
            }
            #pragma unroll
            for (int t = 0; t < 4; ++t) {
                v16bf bfr;
                #pragma unroll
                for (int v = 0; v < 8; ++v) {
                    int k = f * 32 + 2 * v + 16 * hi;        // key index
                    bfr[2 * v]     = Vt[t * 16 + m16][k];    // = V[key=k][d=n]
                    bfr[2 * v + 1] = Vt[t * 16 + m16][k + 1];
                }
                acc[t] = wmma_bf16(pa, bfr, acc[t]);
            }
        }
        __syncthreads();   // protects Vt and the K buffer being refilled next iter
    }

    // ---- normalize and store y[b][l][h*64+d] (f32) ----
    const int b = bh / HH, h = bh % HH;
    #pragma unroll
    for (int t = 0; t < 4; ++t) {
        int d = t * 16 + m16;
        #pragma unroll
        for (int v = 0; v < 8; ++v) {
            int l = q0 + wave * 16 + v + 8 * hi;
            y[((size_t)(b * LL + l)) * EE + h * DD + d] = acc[t][v] / lrun[v];
        }
    }
}

// ---------------------------------------------------------------------------
extern "C" void kernel_launch(void* const* d_in, const int* in_sizes, int n_in,
                              void* d_out, int out_size, void* d_ws, size_t ws_size,
                              hipStream_t stream) {
    const float* x  = (const float*)d_in[0];
    const float* Wq = (const float*)d_in[1];
    const float* Wk = (const float*)d_in[2];
    const float* Wv = (const float*)d_in[3];
    const float* bq = (const float*)d_in[4];
    const float* bk = (const float*)d_in[5];
    const float* bv = (const float*)d_in[6];
    // d_in[7] = mask: causality is computed analytically in-kernel.

    const size_t QSZ = (size_t)BB * HH * LL * DD;   // elements per Q/K/V buffer
    __bf16* Qb = (__bf16*)d_ws;
    __bf16* Kb = Qb + QSZ;
    __bf16* Vb = Kb + QSZ;

    dim3 g1((BB * LL) / 64, EE / 64, 3);            // (128, 12, 3)
    qkv_gemm_1250<<<g1, 128, 0, stream>>>(x, Wq, Wk, Wv, bq, bk, bv, Qb, Kb, Vb);

    dim3 g2(LL / 64, BB * HH);                      // (32, 48)
    attn_flash_1250<<<g2, 128, 0, stream>>>(Qb, Kb, Vb, (float*)d_out);
}
